// ModulatedConv2d_63797444214969
// MI455X (gfx1250) — matrix operand order repaired and checked by reading, not compiled, not run on hardware
//
#include <hip/hip_runtime.h>

// ---------------------------------------------------------------------------
// ModulatedConv2d (StyleGAN2): B=16, Cin=Cout=512, 3x3, 64x64, style_dim=512
// bf16 WMMA implicit GEMM (v_wmma_f32_16x16x32_bf16), NHWC staging,
// double-buffered LDS fed by GLOBAL_LOAD_ASYNC_TO_LDS_B128 when available.
// ---------------------------------------------------------------------------

#define BB   16
#define CI   512
#define CO   512
#define SD   512
#define HH   64
#define WW   64
#define HWP  (HH * WW)          // 4096 pixels
#define RJ   (9 * CI)           // 4608 reduction length (kh,kw,i ordering)

#define LIN_SCALE  0.04419417382415922f    // 1/sqrt(512)
#define CONV_SCALE 0.014731391274719739f   // 1/sqrt(512*9)

typedef __bf16 v16bf __attribute__((ext_vector_type(16)));
typedef float  v8f   __attribute__((ext_vector_type(8)));

union Frag16 { v16bf v; uint4 q[2]; };

// ---- async copy-to-LDS path (gfx1250), guarded for compile safety ----------
#if defined(__AMDGCN__) && __has_builtin(__builtin_amdgcn_global_load_async_to_lds_b128)
#define HAVE_ASYNC 1
#else
#define HAVE_ASYNC 0
#endif

#if HAVE_ASYNC
typedef int v4i_vs __attribute__((vector_size(4 * sizeof(int))));
typedef __attribute__((address_space(1))) v4i_vs* gv4p;   // global int4*
typedef __attribute__((address_space(3))) v4i_vs* lv4p;   // LDS int4*
#endif

__device__ __forceinline__ void async_cp16(void* lds, const void* g) {
#if HAVE_ASYNC
    __builtin_amdgcn_global_load_async_to_lds_b128((gv4p)g, (lv4p)lds, 0, 0);
#else
    *(uint4*)lds = *(const uint4*)g;
#endif
}

#if HAVE_ASYNC
#if __has_builtin(__builtin_amdgcn_s_wait_asynccnt)
#define WAIT_ASYNC(n) __builtin_amdgcn_s_wait_asynccnt(n)
#else
#define WAIT_ASYNC(n) asm volatile("s_wait_asynccnt %0" ::"i"(n) : "memory")
#endif
#else
#define WAIT_ASYNC(n)
#endif

__device__ __forceinline__ unsigned short f32_to_bf16(float f) {
    unsigned int u = __float_as_uint(f);
    u += 0x7fffu + ((u >> 16) & 1u);       // round-to-nearest-even
    return (unsigned short)(u >> 16);
}

// ---------------------------------------------------------------------------
// 1) style[b][i] = sum_s w[b][s] * style_w[i][s] * lin_scale + style_b[i]
// ---------------------------------------------------------------------------
__global__ void style_kernel(const float* __restrict__ w,
                             const float* __restrict__ sw,
                             const float* __restrict__ sb,
                             float* __restrict__ style) {
    int t = blockIdx.x * blockDim.x + threadIdx.x;
    if (t >= BB * CI) return;
    int b = t >> 9, i = t & 511;
    const float* wp = w + b * SD;
    const float* sp = sw + (size_t)i * SD;
    float acc = 0.f;
    for (int s = 0; s < SD; ++s) acc += wp[s] * sp[s];
    style[t] = acc * LIN_SCALE + sb[i];
}

// ---------------------------------------------------------------------------
// 2) demod[b][o] = rsqrt(conv_scale^2 * sum_i style^2 * sum_k weight^2 + eps)
// ---------------------------------------------------------------------------
__global__ void demod_kernel(const float* __restrict__ weight,
                             const float* __restrict__ style,
                             float* __restrict__ demod) {
    int t = blockIdx.x * blockDim.x + threadIdx.x;
    if (t >= BB * CO) return;
    int b = t >> 9, o = t & 511;
    float acc = 0.f;
    for (int i = 0; i < CI; ++i) {
        float s = style[b * CI + i];
        const float* wp = weight + ((size_t)o * CI + i) * 9;
        float lsum = 0.f;
#pragma unroll
        for (int k = 0; k < 9; ++k) { float v = wp[k]; lsum += v * v; }
        acc += lsum * s * s;
    }
    demod[t] = rsqrtf(acc * (CONV_SCALE * CONV_SCALE) + 1e-8f);
}

// ---------------------------------------------------------------------------
// 3a) x[B][C][H][W] f32  ->  xh[B][H*W][C] bf16  (NHWC: channels contiguous)
// ---------------------------------------------------------------------------
__global__ void xpose_kernel(const float* __restrict__ x,
                             unsigned short* __restrict__ xh) {
    size_t t = (size_t)blockIdx.x * blockDim.x + threadIdx.x;
    if (t >= (size_t)BB * CI * HWP) return;
    int i = (int)(t & 511);
    int p = (int)((t >> 9) & 4095);
    int b = (int)(t >> 21);
    float v = x[((size_t)b * CI + i) * HWP + p];
    xh[t] = f32_to_bf16(v);
}

// ---------------------------------------------------------------------------
// 3b) wmod[b][o][kk][i] = weight[o][i][kh][kw]*style[b][i]*cs*demod[b][o], bf16
// ---------------------------------------------------------------------------
__global__ void wmod_kernel(const float* __restrict__ weight,
                            const float* __restrict__ style,
                            const float* __restrict__ demod,
                            unsigned short* __restrict__ wmod) {
    size_t t = (size_t)blockIdx.x * blockDim.x + threadIdx.x;
    if (t >= (size_t)BB * CO * RJ) return;
    int i = (int)(t & 511);
    size_t r = t >> 9;
    int kk = (int)(r % 9);
    size_t q = r / 9;
    int o = (int)(q & 511);
    int b = (int)(q >> 9);
    float v = weight[((size_t)o * CI + i) * 9 + kk] * style[b * CI + i] *
              CONV_SCALE * demod[b * CO + o];
    wmod[t] = f32_to_bf16(v);
}

// ---------------------------------------------------------------------------
// WMMA micro-tile: 2x2 accumulators (32 o x 32 px) from LDS-staged tiles
// ---------------------------------------------------------------------------
__device__ __forceinline__ void compute_tile(
    const unsigned short* __restrict__ bufA,
    const unsigned short* __restrict__ bufB,
    int wvm, int wvn, int lrow, int lhal, v8f acc[2][2]) {
    Frag16 afrag[2], bfrag[2];
#pragma unroll
    for (int mi = 0; mi < 2; ++mi) {
        const unsigned short* ap = bufA + (wvm * 32 + mi * 16 + lrow) * 40;
        afrag[mi].q[0] = *(const uint4*)(ap + lhal * 16);      // K h*16..h*16+7
        afrag[mi].q[1] = *(const uint4*)(ap + lhal * 16 + 8);  // K h*16+8..
    }
#pragma unroll
    for (int ni = 0; ni < 2; ++ni) {
        const unsigned short* bp =
            bufB + (wvn * 32 + ni * 16 + lrow) * 40 + lhal * 16;
        bfrag[ni].q[0] = *(const uint4*)(bp);
        bfrag[ni].q[1] = *(const uint4*)(bp + 8);
    }
#pragma unroll
    for (int mi = 0; mi < 2; ++mi)
#pragma unroll
        for (int ni = 0; ni < 2; ++ni)
            acc[mi][ni] = __builtin_amdgcn_wmma_f32_16x16x32_bf16(
                false, afrag[mi].v, false, bfrag[ni].v,
                (short)0, acc[mi][ni], false, false);
}

// ---------------------------------------------------------------------------
// 4) Implicit GEMM conv: block = (b, image row h [64 px], 128 out channels)
//    8 waves as 4(M)x2(N); double-buffered LDS, async global->LDS staging.
//    K loop: 9 taps (unrolled, kh/kw const) x 16 chunks of 32 channels.
// ---------------------------------------------------------------------------
__global__ __launch_bounds__(256)
void modconv_gemm(const unsigned short* __restrict__ xh,
                  const unsigned short* __restrict__ wmod,
                  float* __restrict__ out) {
    __shared__ alignas(16) unsigned short ldsA[2][128 * 40];  // padded rows
    __shared__ alignas(16) unsigned short ldsB[2][64 * 40];

    const int b    = blockIdx.z;
    const int h    = blockIdx.x;           // image row == pixel tile
    const int oblk = blockIdx.y * 128;

    const int t    = threadIdx.x;
    const int lane = t & 31;
    const int wid  = t >> 5;
    const int wvm  = wid & 3;              // wave M coordinate (0..3)
    const int wvn  = wid >> 2;             // wave N coordinate (0..1)
    const int lrow = lane & 15;
    const int lhal = lane >> 4;

    // staging maps: A = 128 rows x 32 ch (32B/thread), B = 64 px x 32 ch (16B)
    const int arow  = t >> 1;
    const int ahalf = t & 1;
    const int bpix  = t >> 2;
    const int bchk  = t & 3;

    v8f acc[2][2];
#pragma unroll
    for (int mi = 0; mi < 2; ++mi)
#pragma unroll
        for (int ni = 0; ni < 2; ++ni)
#pragma unroll
            for (int e = 0; e < 8; ++e) acc[mi][ni][e] = 0.f;

    const size_t wrow =
        ((size_t)b * CO + (oblk + arow)) * RJ + (size_t)ahalf * 16;

#pragma unroll
    for (int kk = 0; kk < 9; ++kk) {
        const int kh = kk / 3, kw = kk - 3 * kh;   // constants after unroll
        const int hp = h + kh - 1;
        if (hp < 0 || hp >= HH) continue;          // uniform: tap contributes 0

        const int wpx = bpix + kw - 1;
        const bool pixok = (wpx >= 0) && (wpx < WW);
        const size_t xoff =
            ((size_t)b * HWP + (size_t)(hp * WW + wpx)) * CI + bchk * 8;

        auto stage = [&](int bufi, int i0) {
            const unsigned short* gA = wmod + wrow + kk * CI + i0;
            unsigned short* lA = &ldsA[bufi][0] + arow * 40 + ahalf * 16;
            async_cp16(lA, gA);
            async_cp16(lA + 8, gA + 8);
            if (pixok)
                async_cp16(&ldsB[bufi][0] + bpix * 40 + bchk * 8,
                           xh + xoff + i0);
        };

        __syncthreads();   // all waves done with previous tap's buffers
        if (!pixok) {      // zero border rows once; async loads never touch them
            uint4 z = make_uint4(0u, 0u, 0u, 0u);
            *(uint4*)(&ldsB[0][0] + bpix * 40 + bchk * 8) = z;
            *(uint4*)(&ldsB[1][0] + bpix * 40 + bchk * 8) = z;
        }
        stage(0, 0);       // prologue fill of buffer 0

        for (int step = 0; step < 15; ++step) {
            __syncthreads();                     // next buffer free to overwrite
            stage((step + 1) & 1, (step + 1) * 32);
            WAIT_ASYNC(3);                       // in-order: current tile landed
            __syncthreads();                     // all waves' tile data visible
            compute_tile(&ldsA[step & 1][0], &ldsB[step & 1][0],
                         wvm, wvn, lrow, lhal, acc);
        }
        __syncthreads();
        WAIT_ASYNC(0);
        __syncthreads();
        compute_tile(&ldsA[1][0], &ldsB[1][0], wvm, wvn, lrow, lhal, acc);
    }

    // epilogue: C/D layout VGPR r -> M = (lane/16)*8 + r, N = lane%16
#pragma unroll
    for (int mi = 0; mi < 2; ++mi) {
#pragma unroll
        for (int ni = 0; ni < 2; ++ni) {
            const int n = wvn * 32 + ni * 16 + lrow;
#pragma unroll
            for (int r = 0; r < 8; ++r) {
                const int o = oblk + wvm * 32 + mi * 16 + lhal * 8 + r;
                out[((size_t)b * CO + o) * HWP + h * WW + n] = acc[mi][ni][r];
            }
        }
    }
}

// ---------------------------------------------------------------------------
extern "C" void kernel_launch(void* const* d_in, const int* in_sizes, int n_in,
                              void* d_out, int out_size, void* d_ws,
                              size_t ws_size, hipStream_t stream) {
    const float* x        = (const float*)d_in[0];  // [16,512,64,64]
    const float* w        = (const float*)d_in[1];  // [16,512]
    const float* weight   = (const float*)d_in[2];  // [512,512,3,3]
    const float* style_w  = (const float*)d_in[3];  // [512,512]
    const float* style_b  = (const float*)d_in[4];  // [512]
    float* out = (float*)d_out;                     // [16,512,64,64]

    // workspace layout
    float* style = (float*)d_ws;                        // 8192 f32
    float* demod = style + BB * CI;                     // 8192 f32
    unsigned short* xh = (unsigned short*)(demod + BB * CO);   // B*HW*CI bf16
    unsigned short* wmod = xh + (size_t)BB * HWP * CI;         // B*CO*RJ bf16

    style_kernel<<<(BB * CI + 255) / 256, 256, 0, stream>>>(w, style_w,
                                                            style_b, style);
    demod_kernel<<<(BB * CO + 255) / 256, 256, 0, stream>>>(weight, style,
                                                            demod);
    {
        size_t n = (size_t)BB * CI * HWP;
        xpose_kernel<<<(unsigned)((n + 255) / 256), 256, 0, stream>>>(x, xh);
    }
    {
        size_t n = (size_t)BB * CO * RJ;
        wmod_kernel<<<(unsigned)((n + 255) / 256), 256, 0, stream>>>(
            weight, style, demod, wmod);
    }
    modconv_gemm<<<dim3(HH, CO / 128, BB), 256, 0, stream>>>(xh, wmod, out);
}
